// Conv3DMod_42176578846780
// MI455X (gfx1250) — compile-verified, matrix operand order checked by reading, not run on hardware
//
#include <hip/hip_runtime.h>

typedef __attribute__((ext_vector_type(16))) _Float16 v16h;
typedef __attribute__((ext_vector_type(8)))  _Float16 v8h;
typedef __attribute__((ext_vector_type(8)))  float    v8f;

// async-to-LDS builtin operand types: vector-of-4-int pointers with explicit
// address spaces (global=1, LDS=3), per the hipcc diagnostic / builtin def.
typedef int v4i __attribute__((vector_size(16)));
typedef __attribute__((address_space(1))) v4i v4i_g;
typedef __attribute__((address_space(3))) v4i v4i_l;

static constexpr int Bn   = 8;
static constexpr int CIn  = 128;
static constexpr int COu  = 128;
static constexpr int Ktap = 27;            // 3*3*3
static constexpr int Dd = 32, Hh = 32, Ww = 32;
static constexpr int SP  = Dd * Hh * Ww;   // 32768 positions
static constexpr int KT  = Ktap * CIn;     // 3456 GEMM-K extent
static constexpr int Dp  = 34;             // padded (halo) extent
static constexpr int PSP = Dp * Dp * Dp;   // 39304 padded positions

#define ASYNC_LDS __has_builtin(__builtin_amdgcn_global_load_async_to_lds_b128)

// ---------------------------------------------------------------------------
// Kernel 0: zero the halo-padded f16 activation buffer (16B per thread).
// grid*block*8 halfs == Bn*PSP*CIn exactly (19652*256*8).
// ---------------------------------------------------------------------------
__global__ __launch_bounds__(256)
void zero_kernel(v8h* __restrict__ p) {
  const size_t gid = (size_t)blockIdx.x * 256 + threadIdx.x;
  v8h z = {};
  p[gid] = z;
}

// ---------------------------------------------------------------------------
// Kernel 1: modulate + demodulate weights -> f16, K-contiguous layout
//   wA[b][o][tap*128 + cin], row stride KT (6912 B, 16B aligned).
// ---------------------------------------------------------------------------
__global__ __launch_bounds__(256)
void modw_kernel(const float* __restrict__ weight,   // [COu][CIn][27]
                 const float* __restrict__ y,        // [Bn][CIn]
                 _Float16* __restrict__ wA) {
  __shared__ float red[256];
  const int bo = blockIdx.x;
  const int b = bo >> 7;
  const int o = bo & 127;
  const int t = threadIdx.x;

  float partial = 0.0f;
  for (int idx = t; idx < KT; idx += 256) {
    const int tap = idx >> 7;                  // idx = tap*128 + i
    const int i   = idx & 127;
    const float wv = weight[(o * CIn + i) * Ktap + tap];
    const float m  = y[b * CIn + i] + 1.0f;
    const float v  = wv * m;
    partial += v * v;
  }
  red[t] = partial;
  __syncthreads();
  for (int s = 128; s > 0; s >>= 1) {
    if (t < s) red[t] += red[t + s];
    __syncthreads();
  }
  const float dem = rsqrtf(red[0] + 1e-8f);

  _Float16* dst = wA + (size_t)bo * KT;
  for (int idx = t; idx < KT; idx += 256) {
    const int tap = idx >> 7;
    const int i   = idx & 127;
    const float wv = weight[(o * CIn + i) * Ktap + tap];
    const float m  = y[b * CIn + i] + 1.0f;
    dst[idx] = (_Float16)(wv * m * dem);
  }
}

// ---------------------------------------------------------------------------
// Kernel 2: x f32 [b][cin][32^3] -> f16 halo-padded [b][34^3][cin]
// (cin contiguous = GEMM-K; halo stays zero from zero_kernel)
// ---------------------------------------------------------------------------
__global__ __launch_bounds__(256)
void xcast_kernel(const float* __restrict__ x, _Float16* __restrict__ xh) {
  const size_t gid = (size_t)blockIdx.x * 256 + threadIdx.x;
  const int b   = (int)(gid / (size_t)(CIn * SP));
  const int rem = (int)(gid % (size_t)(CIn * SP));
  const int c = rem / SP;
  const int p = rem % SP;
  const int dz = p >> 10;
  const int hy = (p >> 5) & 31;
  const int wx = p & 31;
  const size_t ppos = ((size_t)(dz + 1) * Dp + (hy + 1)) * Dp + (wx + 1);
  const float v = __builtin_nontemporal_load(x + gid);
  xh[((size_t)b * PSP + ppos) * CIn + c] = (_Float16)v;
}

// ---------------------------------------------------------------------------
// Kernel 3: implicit-GEMM conv via WMMA f32_16x16x32_f16.
// 256 threads = 8 waves; WG owns (b, d, h, 16-wide w strip); wave owns a
// 16-row C_out tile. B tile (9 rows x 18 pos x 128 cin f16 = 41.5 KB) is
// staged once per WG into LDS via async-to-LDS, shared by all 8 waves.
// K loop: 27 taps x 4 cin-chunks -> 108 WMMAs per wave, branch-free.
// ---------------------------------------------------------------------------
__global__ __launch_bounds__(256)
void conv_kernel(const _Float16* __restrict__ xh,  // [Bn][PSP][CIn] f16 (haloed)
                 const _Float16* __restrict__ wA,  // [Bn*COu][KT]   f16
                 float* __restrict__ out) {        // [Bn][COu][SP]  f32
  __shared__ __align__(16) _Float16 tile[9 * 18 * CIn];  // 41472 B

  const int blk  = blockIdx.x;
  const int wblk = blk & 1;
  const int h    = (blk >> 1) & 31;
  const int d    = (blk >> 6) & 31;
  const int b    = blk >> 11;
  const int w0   = wblk * 16;
  const int tid  = threadIdx.x;

  // ---- cooperative stage: 9 rows x (18*128 halfs = 288 x 16B chunks) ----
  for (int idx = tid; idx < 9 * 288; idx += 256) {
    const int row   = idx / 288;            // kd*3 + kh
    const int chunk = idx % 288;
    const int kd = row / 3, kh = row % 3;
    const size_t goff =
        ((size_t)b * PSP +
         (size_t)(((d + kd) * Dp + (h + kh)) * Dp + w0)) * (size_t)CIn +
        (size_t)chunk * 8;
    _Float16* lp = tile + (size_t)idx * 8;
#if ASYNC_LDS
    __builtin_amdgcn_global_load_async_to_lds_b128(
        (v4i_g*)(xh + goff), (v4i_l*)lp, 0, 0);
#else
    *(v8h*)lp = *(const v8h*)(xh + goff);
#endif
  }
#if ASYNC_LDS
#if __has_builtin(__builtin_amdgcn_s_wait_asynccnt)
  __builtin_amdgcn_s_wait_asynccnt(0);
#else
  asm volatile("s_wait_asynccnt 0x0" ::: "memory");
#endif
#endif
  __syncthreads();

  // ---- per-wave GEMM ----
  const int wave = tid >> 5;
  const int lane = tid & 31;
  const int n    = lane & 15;        // A: row M; B: col N
  const bool loh = lane < 16;
  const int aoff = loh ? 0 : 8;      // ISA 7.12.2 16-bit A/B lane->K mapping
  const int o0   = wave * 16;

  const _Float16* arow = wA + (size_t)(b * COu + o0 + n) * KT;

  v8f acc = {};
#pragma unroll
  for (int kd = 0; kd < 3; ++kd) {
#pragma unroll
    for (int kh = 0; kh < 3; ++kh) {
      const _Float16* trow = tile + (kd * 3 + kh) * 18 * CIn;
#pragma unroll
      for (int kw = 0; kw < 3; ++kw) {
        const _Float16* bcol = trow + (kw + n) * CIn;   // per-lane column
        const int kbase = ((kd * 3 + kh) * 3 + kw) * CIn;
#pragma unroll
        for (int cc = 0; cc < 4; ++cc) {
          const int kk = kbase + cc * 32;
          union { v16h v; v8h h2[2]; } ua, ub;
          ua.h2[0] = *(const v8h*)(arow + kk + aoff);
          ua.h2[1] = *(const v8h*)(arow + kk + aoff + 16);
          ub.h2[0] = *(const v8h*)(bcol + cc * 32 + aoff);
          ub.h2[1] = *(const v8h*)(bcol + cc * 32 + aoff + 16);
          acc = __builtin_amdgcn_wmma_f32_16x16x32_f16(
              /*neg_a=*/false, ua.v, /*neg_b=*/false, ub.v,
              /*c_mod=*/(short)0, acc, /*reuse_a=*/false, /*reuse_b=*/false);
        }
      }
    }
  }

  // C/D layout: VGPR r -> M = r (lanes 0-15) / r+8 (lanes 16-31); N = lane%16
  const int mbase = o0 + (loh ? 0 : 8);
  float* obase = out + (size_t)(b * COu) * SP +
                 (size_t)(d * 1024 + h * 32 + w0 + n);
#pragma unroll
  for (int r = 0; r < 8; ++r) {
    __builtin_nontemporal_store(acc[r], obase + (size_t)(mbase + r) * SP);
  }
}

// ---------------------------------------------------------------------------
extern "C" void kernel_launch(void* const* d_in, const int* in_sizes, int n_in,
                              void* d_out, int out_size, void* d_ws, size_t ws_size,
                              hipStream_t stream) {
  const float* x      = (const float*)d_in[0];  // [8,128,32,32,32]
  const float* y      = (const float*)d_in[1];  // [8,128]
  const float* weight = (const float*)d_in[2];  // [128,128,3,3,3]
  float* out = (float*)d_out;                   // [8,128,32,32,32]

  _Float16* wA = (_Float16*)d_ws;                           // 7.1 MB
  _Float16* xh = (_Float16*)((char*)d_ws + (8ull << 20));   // 80.5 MB (haloed)

  // zero halo buffer: Bn*PSP*CIn halfs / 8 per thread / 256 per block
  zero_kernel<<<(Bn * PSP * CIn) / (8 * 256), 256, 0, stream>>>((v8h*)xh);
  modw_kernel<<<Bn * COu, 256, 0, stream>>>(weight, y, wA);
  xcast_kernel<<<(Bn * CIn * SP) / 256, 256, 0, stream>>>(x, xh);
  conv_kernel<<<Bn * Dd * Hh * (Ww / 16), 256, 0, stream>>>(xh, wA, out);
}
